// ResidueMPNN_89644557403156
// MI455X (gfx1250) — matrix operand order repaired and checked by reading, not compiled, not run on hardware
//
#include <hip/hip_runtime.h>
#include <hip/hip_bf16.h>

// ---------------------------------------------------------------------------
// ResidueMPNN for MI455X (gfx1250, wave32, WMMA).
//
// Algebraic optimization: rdp = residue_diff @ conv_wrd is only used for
// a_r = (rdp * ard).sum(-1).  Fold:  a_r_e = t[dst]-t[src]+cB, where
//   M_l[c,h] = sum_c' wrd[c, h*C+c'] * ard[h,c']        (128x4, tiny)
//   t        = (z_res @ w_rdiff) @ M_l                  ([N,4])
//   cB[h]    = b_rdiff . M_l[:,h]
// eliminating the 160000x128x512 GEMM and its 327MB intermediate.
//
// GEMM: block = 8 waves = 80 rows x 128 cols.  A tile staged in LDS as f16
// (converted once, cooperatively); each lane's A fragment = 2x ds_load_b128.
// Each wave holds 5 accumulators (5 M-subtiles) and issues 5 back-to-back
// v_wmma_f32_16x16x32_f16 per K-step, reusing one 32B B-fragment load from
// weights pre-packed into the per-lane WMMA-B layout.
// ---------------------------------------------------------------------------

typedef __attribute__((ext_vector_type(16))) _Float16 v16h;
typedef __attribute__((ext_vector_type(8)))  _Float16 v8h;
typedef __attribute__((ext_vector_type(8)))  float    v8f;

#define EPI_NONE   0
#define EPI_BIAS   1
#define EPI_LN_ELU 2
#define EPI_RES_LN 3

#define MT 5            // 16-row M-subtiles per block (80 rows)
#define MROWS (MT * 16)

static __device__ __forceinline__ float wave_sum(float v) {
  for (int off = 16; off; off >>= 1) v += __shfl_xor(v, off, 32);
  return v;
}

static __device__ __forceinline__ void atomicMaxF(float* addr, float v) {
  if (v >= 0.f) atomicMax((int*)addr, __float_as_int(v));
  else          atomicMin((unsigned int*)addr, __float_as_uint(v));
}

// Pack fp32 weight W[K,N] (row-major) into per-lane WMMA B-fragment layout.
// Tile (kt,nt) -> 32 lanes x 16 halves; lane: n = lane&15, K = 16*(lane>>4)+i.
__global__ void pack_b_kernel(const float* __restrict__ W, _Float16* __restrict__ out,
                              int K, int N) {
  int t = blockIdx.x * blockDim.x + threadIdx.x;
  int NT = N >> 4, KT = K >> 5;
  int total = KT * NT * 32;
  if (t >= total) return;
  int lane = t & 31;
  int tile = t >> 5;
  int nt = tile % NT, kt = tile / NT;
  int n = lane & 15, g = lane >> 4;
  _Float16* o = out + (size_t)t * 16;
  const float* w = W + (size_t)(kt * 32 + 16 * g) * N + nt * 16 + n;
#pragma unroll
  for (int i = 0; i < 16; ++i) o[i] = (_Float16)w[(size_t)i * N];
}

// C[M,Ntot] = A[M,K] @ Bpacked (+bias) with fused epilogue.
// grid = (M/80, Ntot/128), block = 256 (8 waves; wave -> one 16-col tile,
// 5 M-subtiles).  LN epilogues require Ntot == 128 (gridDim.y == 1).
__global__ __launch_bounds__(256) void gemm_wmma_kernel(
    const float* __restrict__ A, const _Float16* __restrict__ Bpk,
    const float* __restrict__ bias, const float* __restrict__ resid,
    const float* __restrict__ lng, const float* __restrict__ lnb,
    float* __restrict__ C, int M, int Ntot, int K, int epi) {
  __shared__ _Float16 shA[MROWS * 32];     // f16 A tile, row-major (5 KB)
  __shared__ float    shC[MROWS * 128];    // epilogue slab (40 KB)
  const int tid = threadIdx.x, lane = tid & 31, wave = tid >> 5;
  const int mblk = blockIdx.x;
  const int ntile = blockIdx.y * 8 + wave;
  const int NT = Ntot >> 4;
  const int mlane = lane & 15, g = lane >> 4;
  const int KT = K >> 5;
  v8f acc[MT];
#pragma unroll
  for (int mt = 0; mt < MT; ++mt) acc[mt] = (v8f){};

  for (int kt = 0; kt < KT; ++kt) {
    // Stage A tile (80 rows x 32 cols): fp32 pair -> f16 pair, once per block.
    // 1280 pairs / 256 threads = 5 per thread; consecutive tids hit
    // consecutive pairs -> coalesced 8B global loads, ds_store_b32.
#pragma unroll
    for (int it = 0; it < MT; ++it) {
      int p = tid + it * 256;
      int row = p >> 4, cp = (p & 15) * 2;
      float2 f = *(const float2*)&A[(size_t)(mblk * MROWS + row) * K + kt * 32 + cp];
      union { _Float16 h[2]; unsigned u; } pk;
      pk.h[0] = (_Float16)f.x;
      pk.h[1] = (_Float16)f.y;
      *(unsigned*)&shA[row * 32 + cp] = pk.u;
    }
    if (kt + 1 < KT)  // hint next tile (global_prefetch_b8)
      __builtin_prefetch(&A[(size_t)(mblk * MROWS + (tid >> 4)) * K + (kt + 1) * 32], 0, 1);
    __syncthreads();
    // One B fragment per wave per K-step, reused by all 5 WMMAs.
    v16h b = *(const v16h*)(Bpk + (((size_t)kt * NT + ntile) * 32 + lane) * 16);
#pragma unroll
    for (int mt = 0; mt < MT; ++mt) {
      // A fragment: halves 0..15 = K[16g..16g+15]
      const _Float16* ar = &shA[(mt * 16 + mlane) * 32];
      v8h lo = *(const v8h*)(ar + 16 * g);
      v8h hi = *(const v8h*)(ar + 16 * g + 8);
      v16h a = __builtin_shufflevector(lo, hi, 0, 1, 2, 3, 4, 5, 6, 7,
                                       8, 9, 10, 11, 12, 13, 14, 15);
      acc[mt] = __builtin_amdgcn_wmma_f32_16x16x32_f16(false, a, false, b,
                                                       (short)0, acc[mt], false, false);
    }
    __syncthreads();
  }
  // C layout: VGPR r -> row r + 8*(lane>>4); col = lane&15 within tile.
  const int gcol = ntile * 16 + mlane;
  if (epi == EPI_NONE || epi == EPI_BIAS) {
    float bv = (epi == EPI_BIAS) ? bias[gcol] : 0.f;
#pragma unroll
    for (int mt = 0; mt < MT; ++mt)
#pragma unroll
      for (int r = 0; r < 8; ++r) {
        int grow = mblk * MROWS + mt * 16 + r + 8 * g;
        C[(size_t)grow * Ntot + gcol] = acc[mt][r] + bv;
      }
  } else {
    float bv = bias ? bias[gcol] : 0.f;
#pragma unroll
    for (int mt = 0; mt < MT; ++mt)
#pragma unroll
      for (int r = 0; r < 8; ++r) {
        int lrow = mt * 16 + r + 8 * g;
        float v = acc[mt][r] + bv;
        if (epi == EPI_RES_LN)
          v += resid[(size_t)(mblk * MROWS + lrow) * 128 + gcol];
        shC[lrow * 128 + gcol] = v;
      }
    __syncthreads();
    // LayerNorm over 128 cols: 8 waves x 10 rows.
#pragma unroll
    for (int rr = 0; rr < MROWS / 8; ++rr) {
      int lrow = wave * (MROWS / 8) + rr;
      float x0 = shC[lrow * 128 + lane];
      float x1 = shC[lrow * 128 + lane + 32];
      float x2 = shC[lrow * 128 + lane + 64];
      float x3 = shC[lrow * 128 + lane + 96];
      float mu = wave_sum(x0 + x1 + x2 + x3) * (1.f / 128.f);
      float d0 = x0 - mu, d1 = x1 - mu, d2 = x2 - mu, d3 = x3 - mu;
      float var = wave_sum(d0 * d0 + d1 * d1 + d2 * d2 + d3 * d3) * (1.f / 128.f);
      float rstd = rsqrtf(var + 1e-5f);
      int grow = mblk * MROWS + lrow;
      float dd[4] = {d0, d1, d2, d3};
#pragma unroll
      for (int q = 0; q < 4; ++q) {
        int col = lane + 32 * q;
        float y = lng[col] * dd[q] * rstd + lnb[col];
        if (epi == EPI_LN_ELU) y = (y > 0.f) ? y : (__expf(y) - 1.f);
        C[(size_t)grow * 128 + col] = y;
      }
    }
  }
}

// M_l[c,h] = sum_c' wrd[c, h*128+c'] * ard[h, c'].   512 threads total.
__global__ void make_M_kernel(const float* __restrict__ wrd,
                              const float* __restrict__ ard,
                              float* __restrict__ Mmat) {
  int t = blockIdx.x * blockDim.x + threadIdx.x;
  if (t >= 512) return;
  int c = t >> 2, h = t & 3;
  float acc = 0.f;
  for (int cc = 0; cc < 128; ++cc)
    acc += wrd[(size_t)c * 512 + h * 128 + cc] * ard[h * 128 + cc];
  Mmat[c * 4 + h] = acc;
}

// cB[h] = b_rdiff . M[:,h].  One wave.
__global__ void make_cB_kernel(const float* __restrict__ b_rdiff,
                               const float* __restrict__ Mmat,
                               float* __restrict__ cB) {
  int lane = threadIdx.x;
#pragma unroll
  for (int h = 0; h < 4; ++h) {
    float p = 0.f;
    for (int c = lane; c < 128; c += 32) p += b_rdiff[c] * Mmat[c * 4 + h];
    p = wave_sum(p);
    if (lane == 0) cB[h] = p;
  }
}

// Per-node: a_s, a_d (attention projections of xp), t = rd2@M, init m/s, zero agg.
__global__ __launch_bounds__(256) void node_attn_kernel(
    const float* __restrict__ xp, const float* __restrict__ rd2,
    const float* __restrict__ asrc, const float* __restrict__ adst,
    const float* __restrict__ Mmat, float* __restrict__ a_s,
    float* __restrict__ a_d, float* __restrict__ t, float* __restrict__ m,
    float* __restrict__ s, float* __restrict__ agg, int Nn) {
  int wave = threadIdx.x >> 5, lane = threadIdx.x & 31;
  int node = blockIdx.x * 8 + wave;
  if (node >= Nn) return;
  const float* xr = xp + (size_t)node * 512;
#pragma unroll
  for (int h = 0; h < 4; ++h) {
    float ps = 0.f, pd = 0.f;
    for (int c = lane; c < 128; c += 32) {
      float xv = xr[h * 128 + c];
      ps += xv * asrc[h * 128 + c];
      pd += xv * adst[h * 128 + c];
    }
    ps = wave_sum(ps);
    pd = wave_sum(pd);
    if (lane == 0) {
      a_s[node * 4 + h] = ps;
      a_d[node * 4 + h] = pd;
      m[node * 4 + h] = -1e30f;
      s[node * 4 + h] = 0.f;
    }
  }
  float th[4] = {0.f, 0.f, 0.f, 0.f};
  for (int c = lane; c < 128; c += 32) {
    float rv = rd2[(size_t)node * 128 + c];
#pragma unroll
    for (int h = 0; h < 4; ++h) th[h] += rv * Mmat[c * 4 + h];
  }
#pragma unroll
  for (int h = 0; h < 4; ++h) {
    th[h] = wave_sum(th[h]);
    if (lane == 0) t[node * 4 + h] = th[h];
  }
  float4 zz = {0.f, 0.f, 0.f, 0.f};
  float4* ag = (float4*)(agg + (size_t)node * 512);
#pragma unroll
  for (int k = 0; k < 4; ++k) ag[k * 32 + lane] = zz;
}

__global__ void edge_alpha_kernel(const int* __restrict__ ei,
                                  const float* __restrict__ a_s,
                                  const float* __restrict__ a_d,
                                  const float* __restrict__ t,
                                  const float* __restrict__ cB,
                                  float* __restrict__ alpha,
                                  float* __restrict__ m, int E) {
  int e = blockIdx.x * blockDim.x + threadIdx.x;
  if (e >= E) return;
  int src = ei[e], dst = ei[E + e];
#pragma unroll
  for (int h = 0; h < 4; ++h) {
    float al = a_s[src * 4 + h] + a_d[dst * 4 + h] + t[dst * 4 + h] -
               t[src * 4 + h] + cB[h];
    al = (al > 0.f) ? al : 0.2f * al;
    alpha[(size_t)e * 4 + h] = al;
    atomicMaxF(&m[dst * 4 + h], al);
  }
}

__global__ void edge_exp_kernel(const int* __restrict__ ei,
                                float* __restrict__ alpha,
                                const float* __restrict__ m,
                                float* __restrict__ s, int E) {
  int e = blockIdx.x * blockDim.x + threadIdx.x;
  if (e >= E) return;
  int dst = ei[E + e];
#pragma unroll
  for (int h = 0; h < 4; ++h) {
    float ev = __expf(alpha[(size_t)e * 4 + h] - m[dst * 4 + h]);
    alpha[(size_t)e * 4 + h] = ev;
    atomicAdd(&s[dst * 4 + h], ev);
  }
}

// Wave per edge: agg[dst] += (alpha/s) * xp[src], 512 channels.
__global__ __launch_bounds__(256) void edge_agg_kernel(
    const int* __restrict__ ei, const float* __restrict__ alpha,
    const float* __restrict__ s, const float* __restrict__ xp,
    float* __restrict__ agg, int E) {
  int wave = threadIdx.x >> 5, lane = threadIdx.x & 31;
  int e = blockIdx.x * 8 + wave;
  if (e >= E) return;
  int src = ei[e], dst = ei[E + e];
  const float4* xs = (const float4*)(xp + (size_t)src * 512);
  float* ad = agg + (size_t)dst * 512;
#pragma unroll
  for (int h = 0; h < 4; ++h) {
    float w = alpha[(size_t)e * 4 + h] / (s[(size_t)dst * 4 + h] + 1e-16f);
    float4 xv = xs[h * 32 + lane];
    int c = h * 128 + lane * 4;
    atomicAdd(ad + c + 0, w * xv.x);
    atomicAdd(ad + c + 1, w * xv.y);
    atomicAdd(ad + c + 2, w * xv.z);
    atomicAdd(ad + c + 3, w * xv.w);
  }
}

__global__ __launch_bounds__(256) void final_out_kernel(
    const float* __restrict__ z, const float* __restrict__ w,
    const float* __restrict__ b, float* __restrict__ out, int Nn) {
  int wave = threadIdx.x >> 5, lane = threadIdx.x & 31;
  int node = blockIdx.x * 8 + wave;
  if (node >= Nn) return;
  float p = 0.f;
  for (int c = lane; c < 128; c += 32) p += z[(size_t)node * 128 + c] * w[c];
  p = wave_sum(p);
  if (lane == 0) out[node] = p + b[0];
}

extern "C" void kernel_launch(void* const* d_in, const int* in_sizes, int n_in,
                              void* d_out, int out_size, void* d_ws, size_t ws_size,
                              hipStream_t stream) {
  (void)in_sizes; (void)n_in; (void)out_size; (void)ws_size;
  const int Nn = 10000, E = 160000, DIN = 1280, C = 128, HC = 512;

  const float* x_seq     = (const float*)d_in[0];
  const float* x_res     = (const float*)d_in[1];
  const int*   ei        = (const int*)d_in[2];
  const float* w_xres    = (const float*)d_in[3];
  const float* b_xres    = (const float*)d_in[4];
  const float* w_rdiff   = (const float*)d_in[5];
  const float* b_rdiff   = (const float*)d_in[6];
  const float* w_xseq    = (const float*)d_in[7];
  const float* b_xseq    = (const float*)d_in[8];
  const float* ln0_g     = (const float*)d_in[9];
  const float* ln0_b     = (const float*)d_in[10];
  const float* conv_wx   = (const float*)d_in[11];
  const float* conv_wrd  = (const float*)d_in[12];
  const float* conv_wagg = (const float*)d_in[13];
  const float* conv_asrc = (const float*)d_in[14];
  const float* conv_adst = (const float*)d_in[15];
  const float* conv_ard  = (const float*)d_in[16];
  const float* conv_bias = (const float*)d_in[17];
  const float* ln_g      = (const float*)d_in[18];
  const float* ln_b      = (const float*)d_in[19];
  const float* w_out     = (const float*)d_in[20];
  const float* b_out     = (const float*)d_in[21];
  float* out = (float*)d_out;

  // ---- workspace carve (bump allocator, 256B aligned) ----
  char* ws = (char*)d_ws;
  size_t off = 0;
  auto alloc = [&](size_t bytes) -> void* {
    off = (off + 255) & ~(size_t)255;
    void* p = ws + off;
    off += bytes;
    return p;
  };
  float* z_res = (float*)alloc((size_t)Nn * C * 4);
  float* rd2   = (float*)alloc((size_t)Nn * C * 4);
  float* z     = (float*)alloc((size_t)Nn * C * 4);
  float* xp    = (float*)alloc((size_t)Nn * HC * 4);
  float* agg   = (float*)alloc((size_t)Nn * HC * 4);
  float* a_s   = (float*)alloc((size_t)Nn * 4 * 4);
  float* a_d   = (float*)alloc((size_t)Nn * 4 * 4);
  float* t     = (float*)alloc((size_t)Nn * 4 * 4);
  float* m     = (float*)alloc((size_t)Nn * 4 * 4);
  float* s     = (float*)alloc((size_t)Nn * 4 * 4);
  float* alpha = (float*)alloc((size_t)E * 4 * 4);
  float* Mmat  = (float*)alloc(512 * 4);
  float* cB    = (float*)alloc(16);
  _Float16* pk_xres  = (_Float16*)alloc((size_t)DIN * C * 2);
  _Float16* pk_xseq  = (_Float16*)alloc((size_t)DIN * C * 2);
  _Float16* pk_rdiff = (_Float16*)alloc((size_t)C * C * 2);
  _Float16* pk_wx[2]   = {(_Float16*)alloc((size_t)C * HC * 2),
                          (_Float16*)alloc((size_t)C * HC * 2)};
  _Float16* pk_wagg[2] = {(_Float16*)alloc((size_t)HC * C * 2),
                          (_Float16*)alloc((size_t)HC * C * 2)};

  auto pack = [&](const float* W, _Float16* P, int K, int N) {
    int total = (K / 32) * (N / 16) * 32;
    pack_b_kernel<<<(total + 255) / 256, 256, 0, stream>>>(W, P, K, N);
  };
  pack(w_xres, pk_xres, DIN, C);
  pack(w_xseq, pk_xseq, DIN, C);
  pack(w_rdiff, pk_rdiff, C, C);
  pack(conv_wx, pk_wx[0], C, HC);
  pack(conv_wx + (size_t)C * HC, pk_wx[1], C, HC);
  pack(conv_wagg, pk_wagg[0], HC, C);
  pack(conv_wagg + (size_t)HC * C, pk_wagg[1], HC, C);

  auto gemm = [&](const float* A, const _Float16* B, const float* bias,
                  const float* resid, const float* lg, const float* lb,
                  float* Cout, int M, int Ntot, int K, int epi) {
    dim3 grid(M / MROWS, Ntot / 128);
    gemm_wmma_kernel<<<grid, 256, 0, stream>>>(A, B, bias, resid, lg, lb,
                                               Cout, M, Ntot, K, epi);
  };

  // residue branch: z_res = x_res @ w_xres + b ; rd2 = z_res @ w_rdiff
  gemm(x_res, pk_xres, b_xres, nullptr, nullptr, nullptr, z_res, Nn, C, DIN, EPI_BIAS);
  gemm(z_res, pk_rdiff, nullptr, nullptr, nullptr, nullptr, rd2, Nn, C, C, EPI_NONE);
  // sequence branch: z = ELU(LN(x_seq @ w_xseq + b))
  gemm(x_seq, pk_xseq, b_xseq, nullptr, ln0_g, ln0_b, z, Nn, C, DIN, EPI_LN_ELU);

  for (int l = 0; l < 2; ++l) {
    gemm(z, pk_wx[l], nullptr, nullptr, nullptr, nullptr, xp, Nn, HC, C, EPI_NONE);
    make_M_kernel<<<2, 256, 0, stream>>>(conv_wrd + (size_t)l * C * HC,
                                         conv_ard + (size_t)l * 4 * C, Mmat);
    make_cB_kernel<<<1, 32, 0, stream>>>(b_rdiff, Mmat, cB);
    node_attn_kernel<<<(Nn + 7) / 8, 256, 0, stream>>>(
        xp, rd2, conv_asrc + (size_t)l * 4 * C, conv_adst + (size_t)l * 4 * C,
        Mmat, a_s, a_d, t, m, s, agg, Nn);
    edge_alpha_kernel<<<(E + 255) / 256, 256, 0, stream>>>(ei, a_s, a_d, t, cB,
                                                           alpha, m, E);
    edge_exp_kernel<<<(E + 255) / 256, 256, 0, stream>>>(ei, alpha, m, s, E);
    edge_agg_kernel<<<(E + 7) / 8, 256, 0, stream>>>(ei, alpha, s, xp, agg, E);
    gemm(agg, pk_wagg[l], conv_bias + (size_t)l * C, z,
         ln_g + (size_t)l * C, ln_b + (size_t)l * C, z, Nn, C, HC, EPI_RES_LN);
  }
  final_out_kernel<<<(Nn + 7) / 8, 256, 0, stream>>>(z, w_out, b_out, out, Nn);
}
